// OptimizedNSA_17987323036181
// MI455X (gfx1250) — matrix operand order repaired and checked by reading, not compiled, not run on hardware
//
#include <hip/hip_runtime.h>

// ---------------- NSA config (matches reference) ----------------
#define SLEN   2048
#define DMODEL 2048
#define NHEAD  16
#define NGRP   4
#define DKV    128
#define NCMPB  127      // compressed blocks
#define NSLCB  32       // selection blocks
#define LBLK   32       // compression block length
#define DSTR_  16       // compression stride
#define LPSZ   64       // selection block size
#define WINSZ  512
#define KLD    (NGRP * DKV)      // 512
#define QLD    (NHEAD * DKV)     // 2048

typedef __attribute__((ext_vector_type(16))) __bf16 v16bf;
typedef __attribute__((ext_vector_type(8)))  __bf16 v8bf;
typedef __attribute__((ext_vector_type(8)))  float  v8f;
typedef int v4i_vs __attribute__((vector_size(16)));   // matches builtin param type

union Frag { v16bf v; v8bf h[2]; };

#define V8F_ZERO ((v8f){0.f,0.f,0.f,0.f,0.f,0.f,0.f,0.f})

__device__ inline __bf16 f2bf(float f) {
  unsigned u = __builtin_bit_cast(unsigned, f);
  u += 0x7FFFu + ((u >> 16) & 1u);               // round-to-nearest-even
  unsigned short s = (unsigned short)(u >> 16);
  return __builtin_bit_cast(__bf16, s);
}
__device__ inline float to_f32(float x) { return x; }
__device__ inline float to_f32(__bf16 x) {
  unsigned short s = __builtin_bit_cast(unsigned short, x);
  unsigned u = ((unsigned)s) << 16;
  return __builtin_bit_cast(float, u);
}
__device__ inline void storeC(float* p, float v) { *p = v; }
__device__ inline void storeC(__bf16* p, float v) { *p = f2bf(v); }

__device__ inline v8f wmma_bf16(const Frag& a, const Frag& b, v8f c) {
  return __builtin_amdgcn_wmma_f32_16x16x32_bf16(
      /*neg_a=*/false, a.v, /*neg_b=*/false, b.v,
      /*c_mod=*/(short)0, c, /*reuse_a=*/false, /*reuse_b=*/false);
}

// ---------------- CDNA5 async global->LDS copy (ASYNCcnt path) ----------------
#if __has_builtin(__builtin_amdgcn_global_load_async_to_lds_b128)
#define HAVE_ASYNC_LDS 1
#else
#define HAVE_ASYNC_LDS 0
#endif

__device__ inline void async_copy_b128(const __bf16* g, __bf16* l) {
#if HAVE_ASYNC_LDS
  __builtin_amdgcn_global_load_async_to_lds_b128(
      (__attribute__((address_space(1))) v4i_vs*)g,
      (__attribute__((address_space(3))) v4i_vs*)l, 0, 0);
#else
  *(v8bf*)l = *(const v8bf*)g;   // fallback: vmem load + ds_store
#endif
}
__device__ inline void wait_async() {
#if HAVE_ASYNC_LDS
#if __has_builtin(__builtin_amdgcn_s_wait_asynccnt)
  __builtin_amdgcn_s_wait_asynccnt(0);
#else
  asm volatile("s_wait_asynccnt 0" ::: "memory");
#endif
#endif
}

// ---------------- Generic tiled GEMM: C[MxN] = A[MxK] * B[KxN] ----------------
// 256 threads = 8 waves (2x4); wave tile 64x32; bf16 WMMA 16x16x32, f32 acc.
template<typename TA, typename TB, typename TC>
__global__ __launch_bounds__(256)
void gemm_wmma(const TA* __restrict__ A, const TB* __restrict__ B,
               TC* __restrict__ C, int M, int N, int K) {
  __shared__ alignas(64) __bf16 As[128 * 48];   // [m][k] stride 48 (96B rows)
  __shared__ alignas(64) __bf16 Bs[128 * 48];   // [n][k] (transposed) stride 48
  const int tid  = threadIdx.x;
  const int wave = tid >> 5, lane = tid & 31;
  const int half = lane >> 4, lm = lane & 15;
  const int wm = wave >> 2, wn = wave & 3;
  const int m0 = blockIdx.y * 128, n0 = blockIdx.x * 128;

  v8f acc[4][2];
#pragma unroll
  for (int i = 0; i < 4; ++i)
#pragma unroll
    for (int j = 0; j < 2; ++j) acc[i][j] = V8F_ZERO;

  const int ar = tid >> 1, ac0 = (tid & 1) * 16;   // A: 128 rows x 32 cols
  const int bk = tid >> 3, bc0 = (tid & 7) * 16;   // B: 32 rows x 128 cols

  for (int k0 = 0; k0 < K; k0 += 32) {             // K is a multiple of 32 here
#pragma unroll
    for (int i = 0; i < 16; ++i) {
      float v = 0.f;
      int gr = m0 + ar;
      if (gr < M) v = to_f32(A[(size_t)gr * K + (k0 + ac0 + i)]);
      As[ar * 48 + ac0 + i] = f2bf(v);
    }
#pragma unroll
    for (int i = 0; i < 16; ++i) {
      float v = 0.f;
      int gc = n0 + bc0 + i;
      if (gc < N) v = to_f32(B[(size_t)(k0 + bk) * N + gc]);
      Bs[(bc0 + i) * 48 + bk] = f2bf(v);
    }
    __syncthreads();

    Frag af[4], bfr[2];
#pragma unroll
    for (int f = 0; f < 4; ++f) {
      const __bf16* p = &As[(wm * 64 + f * 16 + lm) * 48 + half * 16];
      af[f].h[0] = *(const v8bf*)p;
      af[f].h[1] = *(const v8bf*)(p + 8);
    }
#pragma unroll
    for (int f = 0; f < 2; ++f)
      bfr[f].v = *(const v16bf*)&Bs[(wn * 32 + f * 16 + lm) * 48 + half * 16];
#pragma unroll
    for (int i = 0; i < 4; ++i)
#pragma unroll
      for (int j = 0; j < 2; ++j)
        acc[i][j] = wmma_bf16(af[i], bfr[j], acc[i][j]);
    __syncthreads();
  }

#pragma unroll
  for (int i = 0; i < 4; ++i)
#pragma unroll
    for (int j = 0; j < 2; ++j) {
      int col = n0 + wn * 32 + j * 16 + lm;
      if (col >= N) continue;
#pragma unroll
      for (int r = 0; r < 8; ++r) {
        int row = m0 + wm * 64 + i * 16 + half * 8 + r;
        if (row < M) storeC(&C[(size_t)row * N + col], acc[i][j][r]);
      }
    }
}

// ---------------- Gather compression windows: kwin[g*127+j][pos*128+d] ----------------
__global__ void nsa_gather_windows(const __bf16* __restrict__ kbf,
                                   const __bf16* __restrict__ vbf,
                                   __bf16* __restrict__ kwin,
                                   __bf16* __restrict__ vwin) {
  int idx = blockIdx.x * 256 + threadIdx.x;
  if (idx >= NGRP * NCMPB * LBLK * DKV) return;
  int r = idx >> 12, c = idx & 4095;
  int g = r / NCMPB, j = r % NCMPB;
  int pos = c >> 7, d = c & 127;
  size_t src = (size_t)(j * DSTR_ + pos) * KLD + g * DKV + d;
  kwin[idx] = kbf[src];
  vwin[idx] = vbf[src];
}

// ---------------- Pack V transposed (vT[g][d][t]) and vcT[g][d][c] (c=127 padded 0) ----
__global__ void nsa_pack_vT(const __bf16* __restrict__ vbf,
                            const __bf16* __restrict__ vcbf,
                            __bf16* __restrict__ vT,
                            __bf16* __restrict__ vcT) {
  int idx = blockIdx.x * 256 + threadIdx.x;
  if (idx < NGRP * DKV * SLEN) {
    int g = idx >> 18, d = (idx >> 11) & 127, t = idx & 2047;
    vT[idx] = vbf[(size_t)t * KLD + g * DKV + d];
  }
  if (idx < NGRP * DKV * 128) {
    int g = idx >> 14, d = (idx >> 7) & 127, c = idx & 127;
    unsigned short z = 0;
    vcT[idx] = (c < NCMPB) ? vcbf[(size_t)(g * NCMPB + c) * DKV + d]
                           : __builtin_bit_cast(__bf16, z);
  }
}

// ---------------- Branch 1: compressed attention + importance ----------------
// one wave per (head, 16-row query tile); 8 waves per block.
// All 8 waves of a block share the same head/group, so kc (127x128) and vcT
// (128x128) are cooperatively staged into LDS once via async global->LDS DMA.
// Dynamic LDS: kc_s (32KB) + vct_s (32KB) + p-tiles (32KB) = 96KB.
__global__ __launch_bounds__(256)
void nsa_cmp_attn(const __bf16* __restrict__ qbf, const __bf16* __restrict__ kcbf,
                  const __bf16* __restrict__ vcT, float* __restrict__ ocmp,
                  float* __restrict__ imp) {
  extern __shared__ __bf16 smem[];
  __bf16* kc_s  = smem;                   // [128][128], rows 0..126 valid
  __bf16* vct_s = smem + 128 * 128;       // [128][128], col 127 (c=127) pre-zeroed in ws
  __bf16* ptb   = smem + 2 * 128 * 128;   // [8][16*128]

  const int tid = threadIdx.x, wave = tid >> 5, lane = tid & 31;
  const int half = lane >> 4, lm = lane & 15;
  const int tile = blockIdx.x * 8 + wave;           // 2048 tiles
  const int h = tile >> 7, s0 = (tile & 127) << 4, g = h >> 2;
  const float scale = 0.08838834764831845f;

  // ---- cooperative async stage of this group's kc / vcT ----
  {
    const __bf16* kc_g  = kcbf + (size_t)g * NCMPB * DKV;   // 127*128 elems
    const __bf16* vct_g = vcT  + (size_t)g * DKV * 128;     // 128*128 elems
#pragma unroll
    for (int r = 0; r < 8; ++r) {
      int e = (r * 256 + tid) * 8;                          // 8 bf16 = 16B per lane
      if (e < NCMPB * DKV) async_copy_b128(kc_g + e, kc_s + e);
      async_copy_b128(vct_g + e, vct_s + e);
    }
    wait_async();
    __syncthreads();
  }

  // Q fragments (A layout) straight from global bf16
  Frag aq[4];
  const __bf16* qrow = qbf + (size_t)(s0 + lm) * QLD + h * DKV;
#pragma unroll
  for (int ks = 0; ks < 4; ++ks) {
    int kb = ks * 32 + half * 16;
    aq[ks].h[0] = *(const v8bf*)(qrow + kb);
    aq[ks].h[1] = *(const v8bf*)(qrow + kb + 8);
  }

  // scores 16 x 128 (cols 0..126 valid)
  v8f sacc[8];
#pragma unroll
  for (int nt = 0; nt < 8; ++nt) sacc[nt] = V8F_ZERO;
#pragma unroll
  for (int nt = 0; nt < 8; ++nt) {
    int crow = nt * 16 + lm; if (crow > NCMPB - 1) crow = NCMPB - 1;
    const __bf16* kcr = kc_s + (size_t)crow * DKV;
#pragma unroll
    for (int ks = 0; ks < 4; ++ks) {
      Frag b; b.v = *(const v16bf*)(kcr + ks * 32 + half * 16);
      sacc[nt] = wmma_bf16(aq[ks], b, sacc[nt]);
    }
  }

  __bf16* ptw = ptb + wave * (16 * 128);
#pragma unroll
  for (int i = 0; i < 8; ++i) {
    int s = s0 + half * 8 + i;
    float x[8]; float mx = -1e30f;
#pragma unroll
    for (int nt = 0; nt < 8; ++nt) {
      int c = nt * 16 + lm;
      bool ok = (c < NCMPB) && (s >= c * DSTR_ + LBLK - 1);
      x[nt] = ok ? sacc[nt][i] * scale : -1e9f;
      mx = fmaxf(mx, x[nt]);
    }
#pragma unroll
    for (int m = 1; m <= 8; m <<= 1) mx = fmaxf(mx, __shfl_xor(mx, m, 32));
    float sum = 0.f;
#pragma unroll
    for (int nt = 0; nt < 8; ++nt) { x[nt] = __expf(x[nt] - mx); sum += x[nt]; }
#pragma unroll
    for (int m = 1; m <= 8; m <<= 1) sum += __shfl_xor(sum, m, 32);
    float inv = (s >= LBLK - 1) ? (1.f / sum) : 0.f;  // no fully-past block yet -> 0
#pragma unroll
    for (int nt = 0; nt < 8; ++nt) {
      float p = x[nt] * inv;
      int c = nt * 16 + lm;
      ptw[(half * 8 + i) * 128 + c] = f2bf(p);
      if (c < NCMPB && p > 0.f) {
        int i1 = c >> 2, i2 = (c * DSTR_ + LBLK - 1) >> 6;  // overlap map
        float* ip = imp + ((size_t)(g * SLEN + s)) * NSLCB;
        atomicAdd(ip + i1, p);
        if (i2 != i1) atomicAdd(ip + i2, p);
      }
    }
  }

  // o_cmp = P (16x128, col 127 zero) x vc (128x128 via vcT in LDS)
  Frag ap[4];
  const __bf16* prow = ptw + lm * 128;
#pragma unroll
  for (int ks = 0; ks < 4; ++ks) {
    int kb = ks * 32 + half * 16;
    ap[ks].h[0] = *(const v8bf*)(prow + kb);
    ap[ks].h[1] = *(const v8bf*)(prow + kb + 8);
  }
  v8f oacc[8];
#pragma unroll
  for (int nt = 0; nt < 8; ++nt) oacc[nt] = V8F_ZERO;
#pragma unroll
  for (int nt = 0; nt < 8; ++nt) {
    const __bf16* vrow = vct_s + (size_t)(nt * 16 + lm) * 128;
#pragma unroll
    for (int ks = 0; ks < 4; ++ks) {
      Frag b; b.v = *(const v16bf*)(vrow + ks * 32 + half * 16);
      oacc[nt] = wmma_bf16(ap[ks], b, oacc[nt]);
    }
  }
#pragma unroll
  for (int nt = 0; nt < 8; ++nt)
#pragma unroll
    for (int i = 0; i < 8; ++i) {
      int s = s0 + half * 8 + i;
      ocmp[((size_t)h * SLEN + s) * DKV + nt * 16 + lm] = oacc[nt][i];
    }
}

// ---------------- Top-16-of-32 selection -> bitmask per (g, s) ----------------
__global__ void nsa_topk(const float* __restrict__ imp, unsigned* __restrict__ selmask) {
  int idx = blockIdx.x * 256 + threadIdx.x;
  if (idx >= NGRP * SLEN) return;
  int s = idx & (SLEN - 1);
  const float* ip = imp + (size_t)idx * NSLCB;
  int cur = s >> 6;
  float v[NSLCB];
  for (int i = 0; i < NSLCB; ++i) {
    float x = ip[i];
    if (i == 0 || i == cur) x = 1e9f;
    else if (i * LPSZ > s)  x = -1e9f;
    v[i] = x;
  }
  unsigned mask = 0;
  for (int it = 0; it < 16; ++it) {
    int bj = 0; float bv = -1e38f;
    for (int j = 0; j < NSLCB; ++j)
      if (!((mask >> j) & 1u) && v[j] > bv) { bv = v[j]; bj = j; }
    mask |= 1u << bj;
  }
  selmask[idx] = mask;
}

// ---------------- Branches 2/3: flash attention over 64-key tiles ----------------
// MODE 0 = selected-block mask, MODE 1 = sliding window
template<int MODE>
__global__ __launch_bounds__(256)
void nsa_attn(const __bf16* __restrict__ qbf, const __bf16* __restrict__ kbf,
              const __bf16* __restrict__ vT, const unsigned* __restrict__ selmask,
              float* __restrict__ out) {
  __shared__ alignas(64) __bf16 pt[8][16 * 64];
  const int tid = threadIdx.x, wave = tid >> 5, lane = tid & 31;
  const int half = lane >> 4, lm = lane & 15;
  const int tile = blockIdx.x * 8 + wave;
  const int h = tile >> 7, s0 = (tile & 127) << 4, g = h >> 2;
  const float scale = 0.08838834764831845f;

  Frag aq[4];
  const __bf16* qrow = qbf + (size_t)(s0 + lm) * QLD + h * DKV;
#pragma unroll
  for (int ks = 0; ks < 4; ++ks) {
    int kb = ks * 32 + half * 16;
    aq[ks].h[0] = *(const v8bf*)(qrow + kb);
    aq[ks].h[1] = *(const v8bf*)(qrow + kb + 8);
  }

  unsigned selm[8]; unsigned orM = 0xFFFFFFFFu;
  if (MODE == 0) {
    orM = 0;
#pragma unroll
    for (int i = 0; i < 8; ++i) {
      selm[i] = selmask[g * SLEN + s0 + half * 8 + i];
      orM |= selm[i];
    }
    orM |= __shfl_xor(orM, 16, 32);   // OR across the two lane-halves (all 16 rows)
  }

  float mrun[8], lrun[8], alpha[8];
  v8f oa[8];
#pragma unroll
  for (int i = 0; i < 8; ++i) { mrun[i] = -1e30f; lrun[i] = 0.f; }
#pragma unroll
  for (int nt = 0; nt < 8; ++nt) oa[nt] = V8F_ZERO;

  const int ktEnd = s0 >> 6;
  int ktStart = 0;
  if (MODE == 1) ktStart = (s0 > WINSZ - 1) ? ((s0 - (WINSZ - 1)) >> 6) : 0;

  __bf16* ptw = pt[wave];
  for (int kt = ktStart; kt <= ktEnd; ++kt) {
    if (MODE == 0 && !((orM >> kt) & 1u)) continue;   // wave-uniform skip
    const int t0 = kt << 6;

    if (kt < ktEnd) {  // prefetch next K tile (global_prefetch_b8)
      const __bf16* nk = kbf + (size_t)(t0 + 64 + lane) * KLD + g * DKV;
      __builtin_prefetch(nk, 0, 3);
      __builtin_prefetch(kbf + (size_t)(t0 + 96 + lane) * KLD + g * DKV, 0, 3);
    }

    // scores 16 x 64
    v8f sacc[4];
#pragma unroll
    for (int nt = 0; nt < 4; ++nt) sacc[nt] = V8F_ZERO;
#pragma unroll
    for (int nt = 0; nt < 4; ++nt) {
      const __bf16* krow = kbf + (size_t)(t0 + nt * 16 + lm) * KLD + g * DKV;
#pragma unroll
      for (int ks = 0; ks < 4; ++ks) {
        Frag b; b.v = *(const v16bf*)(krow + ks * 32 + half * 16);
        sacc[nt] = wmma_bf16(aq[ks], b, sacc[nt]);
      }
    }

    // online softmax update
#pragma unroll
    for (int i = 0; i < 8; ++i) {
      int s = s0 + half * 8 + i;
      float x[4]; float mx = mrun[i];
#pragma unroll
      for (int nt = 0; nt < 4; ++nt) {
        int t = t0 + nt * 16 + lm;
        bool ok = (t <= s);
        if (MODE == 0) ok = ok && ((selm[i] >> kt) & 1u);
        else           ok = ok && (s - t < WINSZ);
        x[nt] = ok ? sacc[nt][i] * scale : -1e9f;
        mx = fmaxf(mx, x[nt]);
      }
#pragma unroll
      for (int m = 1; m <= 8; m <<= 1) mx = fmaxf(mx, __shfl_xor(mx, m, 32));
      float a = __expf(mrun[i] - mx);
      float sum = 0.f;
#pragma unroll
      for (int nt = 0; nt < 4; ++nt) { x[nt] = __expf(x[nt] - mx); sum += x[nt]; }
#pragma unroll
      for (int m = 1; m <= 8; m <<= 1) sum += __shfl_xor(sum, m, 32);
      lrun[i] = lrun[i] * a + sum;
      mrun[i] = mx;
      alpha[i] = a;
#pragma unroll
      for (int nt = 0; nt < 4; ++nt)
        ptw[(half * 8 + i) * 64 + nt * 16 + lm] = f2bf(x[nt]);
    }
#pragma unroll
    for (int nt = 0; nt < 8; ++nt)
#pragma unroll
      for (int i = 0; i < 8; ++i) oa[nt][i] *= alpha[i];

    // O += P (16x64) x V (64x128 via vT)
    Frag ap[2];
    const __bf16* prow = ptw + lm * 64;
#pragma unroll
    for (int ks = 0; ks < 2; ++ks) {
      int kb = ks * 32 + half * 16;
      ap[ks].h[0] = *(const v8bf*)(prow + kb);
      ap[ks].h[1] = *(const v8bf*)(prow + kb + 8);
    }
#pragma unroll
    for (int nt = 0; nt < 8; ++nt) {
      const __bf16* vrow = vT + (size_t)(g * DKV + nt * 16 + lm) * SLEN + t0;
#pragma unroll
      for (int ks = 0; ks < 2; ++ks) {
        Frag b; b.v = *(const v16bf*)(vrow + ks * 32 + half * 16);
        oa[nt] = wmma_bf16(ap[ks], b, oa[nt]);
      }
    }
  }

#pragma unroll
  for (int nt = 0; nt < 8; ++nt)
#pragma unroll
    for (int i = 0; i < 8; ++i) {
      int s = s0 + half * 8 + i;
      out[((size_t)h * SLEN + s) * DKV + nt * 16 + lm] = oa[nt][i] / lrun[i];
    }
}

// ---------------- Gated combine -> bf16 [S, H*DV] ----------------
__global__ void nsa_combine(const float* __restrict__ gf, const float* __restrict__ ocmp,
                            const float* __restrict__ osel, const float* __restrict__ owin,
                            __bf16* __restrict__ ocomb) {
  int idx = blockIdx.x * 256 + threadIdx.x;
  if (idx >= SLEN * QLD) return;
  int s = idx >> 11, r = idx & 2047, h = r >> 7, d = r & 127;
  const float* gp = gf + (size_t)s * (NHEAD * 3) + h * 3;
  float g0 = 1.f / (1.f + __expf(-gp[0]));
  float g1 = 1.f / (1.f + __expf(-gp[1]));
  float g2 = 1.f / (1.f + __expf(-gp[2]));
  size_t ai = ((size_t)h * SLEN + s) * DKV + d;
  float o = g0 * ocmp[ai] + g1 * osel[ai] + g2 * owin[ai];
  ocomb[(size_t)s * QLD + h * DKV + d] = f2bf(o);
}

// ---------------- Host launcher ----------------
extern "C" void kernel_launch(void* const* d_in, const int* in_sizes, int n_in,
                              void* d_out, int out_size, void* d_ws, size_t ws_size,
                              hipStream_t stream) {
  (void)in_sizes; (void)n_in; (void)out_size; (void)ws_size;
  const float* x   = (const float*)d_in[0];
  const float* Wq  = (const float*)d_in[1];
  const float* Wk  = (const float*)d_in[2];
  const float* Wv  = (const float*)d_in[3];
  const float* Wck = (const float*)d_in[4];
  const float* Wcv = (const float*)d_in[5];
  const float* Wg  = (const float*)d_in[6];
  const float* Wo  = (const float*)d_in[7];

  size_t off = 0;
  char* w = (char*)d_ws;
  auto take = [&](size_t n) { char* p = w + off; off = (off + n + 255) & ~(size_t)255; return p; };

  __bf16* qbf   = (__bf16*)take((size_t)SLEN * QLD * 2);
  __bf16* kbf   = (__bf16*)take((size_t)SLEN * KLD * 2);
  __bf16* vbf   = (__bf16*)take((size_t)SLEN * KLD * 2);
  float*  gf    = (float*) take((size_t)SLEN * NHEAD * 3 * 4);
  __bf16* kwin  = (__bf16*)take((size_t)NGRP * NCMPB * LBLK * DKV * 2);
  __bf16* vwin  = (__bf16*)take((size_t)NGRP * NCMPB * LBLK * DKV * 2);
  __bf16* kcbf  = (__bf16*)take((size_t)NGRP * NCMPB * DKV * 2);
  __bf16* vcbf  = (__bf16*)take((size_t)NGRP * NCMPB * DKV * 2);
  __bf16* vcT   = (__bf16*)take((size_t)NGRP * DKV * 128 * 2);
  __bf16* vT    = (__bf16*)take((size_t)NGRP * DKV * SLEN * 2);
  float*  imp   = (float*) take((size_t)NGRP * SLEN * NSLCB * 4);
  unsigned* selmask = (unsigned*)take((size_t)NGRP * SLEN * 4);
  float*  ocmp  = (float*) take((size_t)NHEAD * SLEN * DKV * 4);
  float*  osel  = (float*) take((size_t)NHEAD * SLEN * DKV * 4);
  float*  owin  = (float*) take((size_t)NHEAD * SLEN * DKV * 4);
  __bf16* ocomb = (__bf16*)take((size_t)SLEN * QLD * 2);

  // projections
  gemm_wmma<float, float, __bf16><<<dim3(16, 16), 256, 0, stream>>>(x, Wq, qbf, SLEN, QLD, DMODEL);
  gemm_wmma<float, float, __bf16><<<dim3(4, 16),  256, 0, stream>>>(x, Wk, kbf, SLEN, KLD, DMODEL);
  gemm_wmma<float, float, __bf16><<<dim3(4, 16),  256, 0, stream>>>(x, Wv, vbf, SLEN, KLD, DMODEL);
  gemm_wmma<float, float, float ><<<dim3(1, 16),  256, 0, stream>>>(x, Wg, gf, SLEN, NHEAD * 3, DMODEL);

  // compression: gather windows then learned pooling GEMMs
  {
    int n = NGRP * NCMPB * LBLK * DKV;
    nsa_gather_windows<<<(n + 255) / 256, 256, 0, stream>>>(kbf, vbf, kwin, vwin);
  }
  gemm_wmma<__bf16, float, __bf16><<<dim3(1, 4), 256, 0, stream>>>(kwin, Wck, kcbf, NGRP * NCMPB, DKV, LBLK * DKV);
  gemm_wmma<__bf16, float, __bf16><<<dim3(1, 4), 256, 0, stream>>>(vwin, Wcv, vcbf, NGRP * NCMPB, DKV, LBLK * DKV);

  // transposed V layouts for WMMA B-fragments
  {
    int n = NGRP * DKV * SLEN;
    nsa_pack_vT<<<(n + 255) / 256, 256, 0, stream>>>(vbf, vcbf, vT, vcT);
  }

  (void)hipMemsetAsync(imp, 0, (size_t)NGRP * SLEN * NSLCB * 4, stream);

  // branch 1 + importance (96KB dynamic LDS: kc + vcT tiles + P staging)
  {
    size_t shmem = (size_t)(2 * 128 * 128 + 8 * 16 * 128) * sizeof(__bf16);
    nsa_cmp_attn<<<256, 256, shmem, stream>>>(qbf, kcbf, vcT, ocmp, imp);
  }

  // top-k selection bitmasks
  nsa_topk<<<(NGRP * SLEN + 255) / 256, 256, 0, stream>>>(imp, selmask);

  // branches 2 and 3
  nsa_attn<0><<<256, 256, 0, stream>>>(qbf, kbf, vT, selmask, osel);
  nsa_attn<1><<<256, 256, 0, stream>>>(qbf, kbf, vT, selmask, owin);

  // gated combine + output projection
  nsa_combine<<<(SLEN * QLD + 255) / 256, 256, 0, stream>>>(gf, ocmp, osel, owin, ocomb);
  gemm_wmma<__bf16, float, float><<<dim3(16, 16), 256, 0, stream>>>(ocomb, Wo, (float*)d_out, SLEN, DMODEL, QLD);
}
